// DilateBlock_6880537608925
// MI455X (gfx1250) — compile-verified, matrix operand order checked by reading, not run on hardware
//
#include <hip/hip_runtime.h>
#include <hip/hip_bf16.h>

// ---------- types matching gfx1250 WMMA builtin signatures ----------
typedef __bf16 bf16;
typedef __attribute__((ext_vector_type(16))) __bf16 v16bf;
typedef __attribute__((ext_vector_type(8)))  __bf16 v8bf;
typedef __attribute__((ext_vector_type(8)))  float  v8f;

#define HW    16384            // 128*128
#define NPIX  131072           // 8*128*128
#define WPB   8                // waves per 256-thread block (wave32)

// ---------------------------------------------------------------------------
// Fragment loader for 16-bit WMMA operands from a row-major bf16 matrix.
// ISA 7.12.2 (16-bit A 16x32): lanes 0-15 -> row m, K = {kb..kb+7, kb+16..kb+23}
//                              lanes16-31 -> row m, K = {kb+8..+15, kb+24..+31}
// B (32x16) uses the mirrored layout with "row" = N; same loader works since
// we store weights row-major as W[o][k] and pass row = o.
// Two 16-byte vector loads per lane (ds_load_b128 / global_load_b128).
// ---------------------------------------------------------------------------
static __device__ __forceinline__
v16bf load_frag_rm(const bf16* __restrict__ base, int row, int ld, int kb, int lane)
{
    const int kh = (lane < 16) ? 0 : 8;
    const bf16* p = base + (size_t)row * ld + kb + kh;
    union { v16bf v; v8bf h[2]; } u;
    u.h[0] = *(const v8bf*)(p);
    u.h[1] = *(const v8bf*)(p + 16);
    return u.v;
}

static __device__ __forceinline__ v8f wmma_bf16(v16bf a, v16bf b, v8f c)
{
    return __builtin_amdgcn_wmma_f32_16x16x32_bf16(
        /*neg_a=*/false, a, /*neg_b=*/false, b,
        /*c_mod=*/(short)0, c, /*reuse_a=*/false, /*reuse_b=*/false);
}

static __device__ __forceinline__ void load16f(const bf16* __restrict__ p, float* out)
{
    v8bf a = *(const v8bf*)(p);
    v8bf b = *(const v8bf*)(p + 8);
#pragma unroll
    for (int i = 0; i < 8; ++i) { out[i] = (float)a[i]; out[8 + i] = (float)b[i]; }
}

static __device__ __forceinline__ void store16bf(bf16* __restrict__ p, const float* v)
{
    v8bf a, b;
#pragma unroll
    for (int i = 0; i < 8; ++i) { a[i] = (bf16)v[i]; b[i] = (bf16)v[8 + i]; }
    *(v8bf*)(p) = a;
    *(v8bf*)(p + 8) = b;
}

// ---------------------------------------------------------------------------
// k0: f32 -> bf16 weight cast (row-major preserved)
// ---------------------------------------------------------------------------
__global__ void k0_cvt(const float* __restrict__ src, bf16* __restrict__ dst, int n)
{
    int i = blockIdx.x * 256 + threadIdx.x;
    if (i < n) dst[i] = (bf16)src[i];
}

// ---------------------------------------------------------------------------
// k1: LN1 + QKV GEMM (96 -> 288), output bf16 pixel-major (pix, 288)
// ---------------------------------------------------------------------------
__global__ __launch_bounds__(256)
void k1_ln_qkv(const float* __restrict__ x,
               const float* __restrict__ ln1w, const float* __restrict__ ln1b,
               const bf16* __restrict__ wq, bf16* __restrict__ qkv)
{
    __shared__ __align__(16) bf16 sY[WPB][16][104];   // 26.6 KB

    const int tid  = threadIdx.x;
    const int wave = tid >> 5, lane = tid & 31;
    const int tile = blockIdx.x * WPB + wave;
    const int p0   = tile * 16;
    const int m    = lane & 15, half = lane >> 4;
    const int p    = p0 + m;
    const int bb   = p >> 14, hw = p & (HW - 1);

    // each lane owns 48 of the 96 channels of its pixel (channel-planar input)
    const float* xp = x + ((size_t)bb * 96 + half * 48) * HW + hw;
    float vals[48];
    float s = 0.f, s2 = 0.f;
#pragma unroll
    for (int c = 0; c < 48; ++c) {
        float v = xp[(size_t)c * HW];
        vals[c] = v; s += v; s2 += v * v;
    }
    s  += __shfl_xor(s, 16, 32);
    s2 += __shfl_xor(s2, 16, 32);
    const float mu   = s * (1.f / 96.f);
    const float rstd = rsqrtf(s2 * (1.f / 96.f) - mu * mu + 1e-5f);
#pragma unroll
    for (int c = 0; c < 48; ++c) {
        int ch = half * 48 + c;
        sY[wave][m][ch] = (bf16)((vals[c] - mu) * rstd * ln1w[ch] + ln1b[ch]);
    }
    __syncthreads();

    const bf16* A = &sY[wave][0][0];
    bf16* outb = qkv + (size_t)p0 * 288;
    const int n  = lane & 15;
    const int mb = (lane < 16) ? 0 : 8;
    for (int j = 0; j < 18; ++j) {
        v8f acc = {};
#pragma unroll
        for (int kb = 0; kb < 96; kb += 32) {
            v16bf a = load_frag_rm(A,  m,          104, kb, lane);
            v16bf b = load_frag_rm(wq, j * 16 + n,  96, kb, lane);
            acc = wmma_bf16(a, b, acc);
        }
#pragma unroll
        for (int r = 0; r < 8; ++r)
            outb[(size_t)(r + mb) * 288 + j * 16 + n] = (bf16)acc[r];
    }
}

// ---------------------------------------------------------------------------
// k2: dilated neighborhood attention (3 dilations x 2 heads of dim 16),
// softmax over 9 taps INCLUDING zero-padded taps (matches nn.Unfold+softmax).
// ---------------------------------------------------------------------------
__global__ __launch_bounds__(256)
void k2_attn(const bf16* __restrict__ qkv, bf16* __restrict__ attn)
{
    const int p  = blockIdx.x * 256 + threadIdx.x;
    const int bb = p >> 14, hw = p & (HW - 1);
    const int yy = hw >> 7, xx = hw & 127;
    const bf16* qp = qkv + (size_t)p * 288;
    bf16* op = attn + (size_t)p * 96;

    for (int d = 0; d < 3; ++d) {
        const int dil = d + 1;
        for (int h = 0; h < 2; ++h) {
            const int co = d * 32 + h * 16;
            float q[16];
            load16f(qp + co, q);

            float logit[9];
#pragma unroll
            for (int t = 0; t < 9; ++t) {
                int ny = yy + (t / 3 - 1) * dil;
                int nx = xx + (t % 3 - 1) * dil;
                float acc = 0.f;
                if ((unsigned)ny < 128u && (unsigned)nx < 128u) {
                    float kv[16];
                    load16f(qkv + ((size_t)(bb * HW + ny * 128 + nx) * 288 + 96 + co), kv);
#pragma unroll
                    for (int e = 0; e < 16; ++e) acc += q[e] * kv[e];
                }
                logit[t] = acc * 0.25f;   // HEAD_DIM^-0.5
            }
            float mx = logit[0];
#pragma unroll
            for (int t = 1; t < 9; ++t) mx = fmaxf(mx, logit[t]);
            float pr[9], sum = 0.f;
#pragma unroll
            for (int t = 0; t < 9; ++t) { pr[t] = __expf(logit[t] - mx); sum += pr[t]; }
            const float inv = 1.f / sum;

            float out[16];
#pragma unroll
            for (int e = 0; e < 16; ++e) out[e] = 0.f;
#pragma unroll
            for (int t = 0; t < 9; ++t) {
                int ny = yy + (t / 3 - 1) * dil;
                int nx = xx + (t % 3 - 1) * dil;
                if ((unsigned)ny < 128u && (unsigned)nx < 128u) {
                    float vv[16];
                    load16f(qkv + ((size_t)(bb * HW + ny * 128 + nx) * 288 + 192 + co), vv);
#pragma unroll
                    for (int e = 0; e < 16; ++e) out[e] += pr[t] * vv[e];
                }
            }
#pragma unroll
            for (int e = 0; e < 16; ++e) out[e] *= inv;
            store16bf(op + co, out);
        }
    }
}

// ---------------------------------------------------------------------------
// k3: proj GEMM + bias + residual(x) -> LDS -> fused LN2
// writes: out1 (f32 residual stream, pixel-major 96), y2 (bf16 normalized)
// ---------------------------------------------------------------------------
__global__ __launch_bounds__(256)
void k3_proj_ln2(const bf16* __restrict__ attn, const bf16* __restrict__ wp,
                 const float* __restrict__ proj_b, const float* __restrict__ x,
                 const float* __restrict__ ln2w, const float* __restrict__ ln2b,
                 float* __restrict__ out1, bf16* __restrict__ y2)
{
    __shared__ __align__(16) float sO[WPB][16][100];  // 51.2 KB

    const int tid  = threadIdx.x;
    const int wave = tid >> 5, lane = tid & 31;
    const int tile = blockIdx.x * WPB + wave;
    const int p0   = tile * 16;
    const int bb   = p0 >> 14, hw0 = p0 & (HW - 1);
    const int m    = lane & 15, half = lane >> 4;
    const int n    = lane & 15;
    const int mb   = (lane < 16) ? 0 : 8;

    const bf16* A = attn + (size_t)p0 * 96;
    for (int j = 0; j < 6; ++j) {
        v8f acc = {};
#pragma unroll
        for (int kb = 0; kb < 96; kb += 32) {
            v16bf a = load_frag_rm(A,  m,          96, kb, lane);
            v16bf b = load_frag_rm(wp, j * 16 + n, 96, kb, lane);
            acc = wmma_bf16(a, b, acc);
        }
        const int c = j * 16 + n;
        const float pb = proj_b[c];
        const float* xc = x + ((size_t)bb * 96 + c) * HW + hw0 + mb;
#pragma unroll
        for (int r = 0; r < 8; ++r)
            sO[wave][r + mb][c] = acc[r] + pb + xc[r];
    }
    __syncthreads();

    // fused LN2 over the 96 channels of each pixel row
    float s = 0.f, s2 = 0.f;
#pragma unroll
    for (int c = 0; c < 48; ++c) {
        float v = sO[wave][m][half * 48 + c];
        s += v; s2 += v * v;
    }
    s  += __shfl_xor(s, 16, 32);
    s2 += __shfl_xor(s2, 16, 32);
    const float mu   = s * (1.f / 96.f);
    const float rstd = rsqrtf(s2 * (1.f / 96.f) - mu * mu + 1e-5f);

    float* o1 = out1 + (size_t)(p0 + m) * 96;
    bf16*  yy = y2   + (size_t)(p0 + m) * 96;
#pragma unroll
    for (int c = 0; c < 48; ++c) {
        int ch = half * 48 + c;
        float v = sO[wave][m][ch];
        o1[ch] = v;
        yy[ch] = (bf16)((v - mu) * rstd * ln2w[ch] + ln2b[ch]);
    }
}

// ---------------------------------------------------------------------------
// k4: fc1 GEMM (96 -> 384) + exact GELU -> h1 bf16 (pixel-major 384)
// ---------------------------------------------------------------------------
__global__ __launch_bounds__(256)
void k4_fc1_gelu(const bf16* __restrict__ y2, const bf16* __restrict__ w1,
                 const float* __restrict__ fc1_b, bf16* __restrict__ h1)
{
    const int tid  = threadIdx.x;
    const int wave = tid >> 5, lane = tid & 31;
    const int tile = blockIdx.x * WPB + wave;
    const int p0   = tile * 16;
    const int m    = lane & 15;
    const int n    = lane & 15;
    const int mb   = (lane < 16) ? 0 : 8;

    const bf16* A = y2 + (size_t)p0 * 96;
    bf16* hb = h1 + (size_t)p0 * 384;
    for (int j = 0; j < 24; ++j) {
        v8f acc = {};
#pragma unroll
        for (int kb = 0; kb < 96; kb += 32) {
            v16bf a = load_frag_rm(A,  m,          96, kb, lane);
            v16bf b = load_frag_rm(w1, j * 16 + n, 96, kb, lane);
            acc = wmma_bf16(a, b, acc);
        }
        const float bias = fc1_b[j * 16 + n];
#pragma unroll
        for (int r = 0; r < 8; ++r) {
            float v = acc[r] + bias;
            float g = 0.5f * v * (1.f + erff(v * 0.70710678118654752f));
            hb[(size_t)(r + mb) * 384 + j * 16 + n] = (bf16)g;
        }
    }
}

// ---------------------------------------------------------------------------
// k5: fc2 GEMM (384 -> 96) + bias + residual -> final (B,C,H,W) f32
// ---------------------------------------------------------------------------
__global__ __launch_bounds__(256)
void k5_fc2_res(const bf16* __restrict__ h1, const bf16* __restrict__ w2,
                const float* __restrict__ fc2_b, const float* __restrict__ out1,
                float* __restrict__ out)
{
    const int tid  = threadIdx.x;
    const int wave = tid >> 5, lane = tid & 31;
    const int tile = blockIdx.x * WPB + wave;
    const int p0   = tile * 16;
    const int bb   = p0 >> 14, hw0 = p0 & (HW - 1);
    const int m    = lane & 15;
    const int n    = lane & 15;
    const int mb   = (lane < 16) ? 0 : 8;

    const bf16* A = h1 + (size_t)p0 * 384;
    for (int j = 0; j < 6; ++j) {
        v8f acc = {};
#pragma unroll
        for (int kb = 0; kb < 384; kb += 32) {
            v16bf a = load_frag_rm(A,  m,           384, kb, lane);
            v16bf b = load_frag_rm(w2, j * 16 + n,  384, kb, lane);
            acc = wmma_bf16(a, b, acc);
        }
        const int c = j * 16 + n;
        const float bias = fc2_b[c];
        float* oc = out + ((size_t)bb * 96 + c) * HW + hw0 + mb;
        const float* r1 = out1 + (size_t)(p0 + mb) * 96 + c;
#pragma unroll
        for (int r = 0; r < 8; ++r)
            oc[r] = acc[r] + bias + r1[(size_t)r * 96];
    }
}

// ---------------------------------------------------------------------------
extern "C" void kernel_launch(void* const* d_in, const int* in_sizes, int n_in,
                              void* d_out, int out_size, void* d_ws, size_t ws_size,
                              hipStream_t stream)
{
    (void)in_sizes; (void)n_in; (void)out_size; (void)ws_size;

    const float* x      = (const float*)d_in[0];
    const float* qkv_w  = (const float*)d_in[1];
    const float* proj_w = (const float*)d_in[2];
    const float* proj_b = (const float*)d_in[3];
    const float* ln1_w  = (const float*)d_in[4];
    const float* ln1_b  = (const float*)d_in[5];
    const float* ln2_w  = (const float*)d_in[6];
    const float* ln2_b  = (const float*)d_in[7];
    const float* fc1_w  = (const float*)d_in[8];
    const float* fc1_b  = (const float*)d_in[9];
    const float* fc2_w  = (const float*)d_in[10];
    const float* fc2_b  = (const float*)d_in[11];
    float* out = (float*)d_out;

    char* ws = (char*)d_ws;
    // bf16 weights
    const size_t OFF_WQ = 0;                                  // 288*96
    const size_t OFF_WP = OFF_WQ + 288 * 96 * sizeof(bf16);   // 96*96
    const size_t OFF_W1 = OFF_WP + 96 * 96 * sizeof(bf16);    // 384*96
    const size_t OFF_W2 = OFF_W1 + 384 * 96 * sizeof(bf16);   // 96*384
    const size_t OFF_QKV  = OFF_W2 + 96 * 384 * sizeof(bf16); // bf16 (NPIX,288)
    const size_t OFF_ATTN = OFF_QKV + (size_t)NPIX * 288 * sizeof(bf16);
    const size_t OFF_H1   = OFF_QKV;                          // alias: qkv+attn dead by k4
    const size_t OFF_OUT1 = OFF_ATTN + (size_t)NPIX * 96 * sizeof(bf16);
    const size_t OFF_Y2   = OFF_OUT1 + (size_t)NPIX * 96 * sizeof(float);

    bf16*  wq   = (bf16*)(ws + OFF_WQ);
    bf16*  wp   = (bf16*)(ws + OFF_WP);
    bf16*  w1   = (bf16*)(ws + OFF_W1);
    bf16*  w2   = (bf16*)(ws + OFF_W2);
    bf16*  qkv  = (bf16*)(ws + OFF_QKV);
    bf16*  attn = (bf16*)(ws + OFF_ATTN);
    bf16*  h1   = (bf16*)(ws + OFF_H1);
    float* out1 = (float*)(ws + OFF_OUT1);
    bf16*  y2   = (bf16*)(ws + OFF_Y2);

    // k0: cast weights to bf16 (kept row-major; fragment loads are 2x16B/lane)
    k0_cvt<<<(288 * 96 + 255) / 256, 256, 0, stream>>>(qkv_w, wq, 288 * 96);
    k0_cvt<<<(96 * 96 + 255) / 256, 256, 0, stream>>>(proj_w, wp, 96 * 96);
    k0_cvt<<<(384 * 96 + 255) / 256, 256, 0, stream>>>(fc1_w, w1, 384 * 96);
    k0_cvt<<<(96 * 384 + 255) / 256, 256, 0, stream>>>(fc2_w, w2, 96 * 384);

    const int TILE_BLOCKS = (NPIX / 16) / WPB;   // 8192 tiles / 8 waves = 1024
    k1_ln_qkv <<<TILE_BLOCKS, 256, 0, stream>>>(x, ln1_w, ln1_b, wq, qkv);
    k2_attn   <<<NPIX / 256, 256, 0, stream>>>(qkv, attn);
    k3_proj_ln2<<<TILE_BLOCKS, 256, 0, stream>>>(attn, wp, proj_b, x, ln2_w, ln2_b, out1, y2);
    k4_fc1_gelu<<<TILE_BLOCKS, 256, 0, stream>>>(y2, w1, fc1_b, h1);
    k5_fc2_res<<<TILE_BLOCKS, 256, 0, stream>>>(h1, w2, fc2_b, out1, out);
}
